// MULTIWAVE_SpecGCN_LSTM_67207648248058
// MI455X (gfx1250) — matrix operand order repaired and checked by reading, not compiled, not run on hardware
//
#include <hip/hip_runtime.h>
#include <hip/hip_bf16.h>

typedef __attribute__((ext_vector_type(16))) __bf16 v16bf;
typedef __attribute__((ext_vector_type(8)))  float  v8f;

#define BB 256
#define DD 28
#define NN 23
#define FF 128
#define OO 64
#define HH 128
#define NB (NN*BB)                       // 5888
#define ADJ_ELEMS ((size_t)BB*DD*NN*NN)  // 3791872
#define ZIN_ELEMS ((size_t)NB*DD*65)     // 10716160

// bf16 weight region (element offsets), base = d_ws + ZIN bytes
#define OFF_W1   0
#define OFF_W2   (OFF_W1 + 128*128)
#define OFF_GW   (OFF_W2 + 128*128)      // [64][128]  (gcn_w transposed)
#define OFF_WIH0 (OFF_GW + 64*128)       // [512][96]  (K padded 65->96)
#define OFF_WHH0 (OFF_WIH0 + 512*96)     // [512][128]
#define OFF_WIH1 (OFF_WHH0 + 512*128)    // [512][128]
#define OFF_WHH1 (OFF_WIH1 + 512*128)    // [512][128]
#define WB_TOTAL (OFF_WHH1 + 512*128)    // 286720 bf16 elems

__device__ __forceinline__ __bf16 f2bf(float f) { return (__bf16)f; }
__device__ __forceinline__ float sigm(float x) { return 1.f / (1.f + __expf(-x)); }

__device__ __forceinline__ v8f zero8() {
  v8f z = {0.f,0.f,0.f,0.f,0.f,0.f,0.f,0.f};
  return z;
}

__device__ __forceinline__ v8f wmma_bf16(v16bf a, v16bf b, v8f c) {
  // D = A(16x32 bf16) * B(32x16 bf16) + C(16x16 f32)
  return __builtin_amdgcn_wmma_f32_16x16x32_bf16(false, a, false, b, (short)0, c,
                                                 false, false);
}

// A-operand fragment (16x32 bf16) from row-major bf16 src[row*ld + k].
// ISA layout: lanes 0-15 row=lane, K in {0..7,16..23}; lanes 16-31 row=lane-16,
// K in {8..15,24..31}; dword j packs K pair (kbase, kbase+1).
__device__ __forceinline__ v16bf load_a_frag(const __bf16* src, int ld, int row0,
                                             int k0, int lane) {
  v16bf a;
  const int m = row0 + (lane & 15);
  const int khi = (lane >> 4) & 1;
#pragma unroll
  for (int j = 0; j < 8; ++j) {
    const int k = k0 + ((j & 3) << 1) + ((j >> 2) << 4) + (khi << 3);
    a[2*j]   = src[m*ld + k];
    a[2*j+1] = src[m*ld + k + 1];
  }
  return a;
}

// B-operand fragment (32x16 bf16); storage is [n][k] (column-of-output major)
// so the two K's of each dword are contiguous. lanes 0-15: col=lane, K {0..7,16..23};
// lanes 16-31: col=lane-16, K {8..15,24..31}.
__device__ __forceinline__ v16bf load_b_frag(const __bf16* src, int ld, int n0,
                                             int k0, int lane) {
  v16bf bv;
  const int n = n0 + (lane & 15);
  const int kb = ((lane >> 4) & 1) << 3;
#pragma unroll
  for (int j = 0; j < 8; ++j) {
    const int k = k0 + kb + ((j & 3) << 1) + ((j >> 2) << 4);
    bv[2*j]   = src[n*ld + k];
    bv[2*j+1] = src[n*ld + k + 1];
  }
  return bv;
}

// A-fragment straight from f32 zin rows (K valid < 65, rest zero)
__device__ __forceinline__ v16bf load_x_frag(const float* zin, int rbase, int d,
                                             int k0, int lane) {
  v16bf a;
  const int r = rbase + (lane & 15);
  const int khi = (lane >> 4) & 1;
  const float* p = zin + ((size_t)r*DD + d)*65;
#pragma unroll
  for (int j = 0; j < 8; ++j) {
    const int k = k0 + ((j & 3) << 1) + ((j >> 2) << 4) + (khi << 3);
    a[2*j]   = (k     < 65) ? f2bf(p[k])     : f2bf(0.f);
    a[2*j+1] = (k + 1 < 65) ? f2bf(p[k + 1]) : f2bf(0.f);
  }
  return a;
}

// ---------------------------------------------------------------------------
// Kernel 1: one-shot weight conversion / repack to bf16
// ---------------------------------------------------------------------------
__global__ __launch_bounds__(256)
void convert_weights(const float* __restrict__ w1, const float* __restrict__ w2,
                     const float* __restrict__ gw, const float* __restrict__ wih0,
                     const float* __restrict__ whh0, const float* __restrict__ wih1,
                     const float* __restrict__ whh1, __bf16* __restrict__ out) {
  int i = blockIdx.x * 256 + threadIdx.x;
  if (i < 128*128) { out[OFF_W1 + i] = f2bf(w1[i]); return; }
  i -= 128*128;
  if (i < 128*128) { out[OFF_W2 + i] = f2bf(w2[i]); return; }
  i -= 128*128;
  if (i < 64*128) {  // gwb[o][k] = gcn_w[k][o]
    const int o = i >> 7, k = i & 127;
    out[OFF_GW + i] = f2bf(gw[k*64 + o]);
    return;
  }
  i -= 64*128;
  if (i < 512*96) {  // wih0 padded K 65->96
    const int g = i / 96, k = i % 96;
    out[OFF_WIH0 + i] = f2bf(k < 65 ? wih0[g*65 + k] : 0.f);
    return;
  }
  i -= 512*96;
  if (i < 512*128) { out[OFF_WHH0 + i] = f2bf(whh0[i]); return; }
  i -= 512*128;
  if (i < 512*128) { out[OFF_WIH1 + i] = f2bf(wih1[i]); return; }
  i -= 512*128;
  if (i < 512*128) { out[OFF_WHH1 + i] = f2bf(whh1[i]); return; }
}

// ---------------------------------------------------------------------------
// Kernel 2: fused graph learner + SpecGCN per (b,d); writes adj and zin
// ---------------------------------------------------------------------------
__global__ __launch_bounds__(256, 1)
void graph_kernel(const float* __restrict__ x, const float* __restrict__ adj_real,
                  const float* __restrict__ infection, const int* __restrict__ day_order,
                  const float* __restrict__ gl_b1, const float* __restrict__ gl_b2,
                  const float* __restrict__ glp, const float* __restrict__ vvec,
                  const __bf16* __restrict__ wsb, float* __restrict__ zin,
                  float* __restrict__ adj_out) {
  __shared__ __bf16 xb[32*128];    // x tile, bf16, rows >= 23 zero
  __shared__ __bf16 n1b[32*128];
  __shared__ __bf16 n2b[32*128];
  __shared__ __bf16 xwT[64*32];    // (x @ gcn_w)^T : [o][k]
  __shared__ __bf16 dadb[32*32];   // normalized adjacency, zero-padded
  __shared__ float  s12s[32*32];
  __shared__ float  tAs[23*24];
  __shared__ float  degs[23];
  __shared__ float  dinvs[23];

  const int bd = blockIdx.x;
  const int b = bd / DD, d = bd % DD;
  const int tid = threadIdx.x;
  const int wave = tid >> 5, lane = tid & 31;
  const int hi = (lane >> 4) & 1, ln = lane & 15;
  const float* xp = x + (size_t)bd * NN * FF;

  // phase 1: x -> bf16 LDS
  for (int i = tid; i < 32*128; i += 256) {
    const int r = i >> 7, c = i & 127;
    xb[i] = f2bf(r < NN ? xp[r*FF + c] : 0.f);
  }
  __syncthreads();

  const __bf16* w1b = wsb + OFF_W1;
  const __bf16* w2b = wsb + OFF_W2;
  const __bf16* gwb = wsb + OFF_GW;

  // phase 2: n1 = tanh(xW1^T+b1), n2 = tanh(xW2^T+b2), xw = x@gcn_w
  // 16+16+8 = 40 WMMA tile jobs over 8 waves
  for (int job = wave; job < 40; job += 8) {
    if (job < 32) {
      const bool first = job < 16;
      const int t = job & 15, mt = t >> 3, nt = t & 7;
      const __bf16* W = first ? w1b : w2b;
      const float* bias = first ? gl_b1 : gl_b2;
      v8f acc = zero8();
#pragma unroll
      for (int kk = 0; kk < 4; ++kk)
        acc = wmma_bf16(load_a_frag(xb, 128, mt*16, kk*32, lane),
                        load_b_frag(W, 128, nt*16, kk*32, lane), acc);
      const float bn = bias[nt*16 + ln];
      __bf16* dst = first ? n1b : n2b;
#pragma unroll
      for (int i = 0; i < 8; ++i) {
        const int m = mt*16 + i + hi*8;
        dst[m*128 + nt*16 + ln] = f2bf(tanhf(acc[i] + bn));
      }
    } else {
      const int t = job - 32, mt = t >> 2, ot = t & 3;
      v8f acc = zero8();
#pragma unroll
      for (int kk = 0; kk < 4; ++kk)
        acc = wmma_bf16(load_a_frag(xb, 128, mt*16, kk*32, lane),
                        load_b_frag(gwb, 128, ot*16, kk*32, lane), acc);
#pragma unroll
      for (int i = 0; i < 8; ++i) {
        const int m = mt*16 + i + hi*8;
        xwT[(ot*16 + ln)*32 + m] = f2bf(acc[i]);
      }
    }
  }
  __syncthreads();

  // phase 3: s12 = n1 @ n2^T  (n2 in [m][f] storage == B [n][k] layout)
  if (wave < 4) {
    const int mt = wave >> 1, nt = wave & 1;
    v8f acc = zero8();
#pragma unroll
    for (int kk = 0; kk < 4; ++kk)
      acc = wmma_bf16(load_a_frag(n1b, 128, mt*16, kk*32, lane),
                      load_b_frag(n2b, 128, nt*16, kk*32, lane), acc);
#pragma unroll
    for (int i = 0; i < 8; ++i)
      s12s[(mt*16 + i + hi*8)*32 + nt*16 + ln] = acc[i];
  }
  __syncthreads();

  // phase 4: adjacency assembly + normalization
  const float* ar = adj_real + (size_t)bd * NN * NN;
  if (tid < NN) {
    float s = 0.f;
    for (int j = 0; j < NN; ++j) s += ar[tid*NN + j];
    degs[tid] = s;
  }
  __syncthreads();
  for (int p = tid; p < NN*NN; p += 256) {
    const int i = p / NN, j = p % NN;
    const float sv = s12s[i*32 + j] - s12s[j*32 + i];
    const float al = fmaxf(0.f, tanhf(sv));
    const float md = sigm(glp[p] * degs[i] * degs[j]);
    const float a = al + md * ar[p];
    adj_out[(size_t)bd*NN*NN + p] = a;
    tAs[i*24 + j] = a + (i == j ? 1.f : 0.f);
  }
  __syncthreads();
  if (tid < NN) {
    float s = 0.f;
    for (int j = 0; j < NN; ++j) s += tAs[tid*24 + j];
    dinvs[tid] = rsqrtf(s);
  }
  __syncthreads();
  for (int p = tid; p < 32*32; p += 256) {
    const int i = p >> 5, j = p & 31;
    const float vv = (i < NN && j < NN) ? dinvs[i]*tAs[i*24 + j]*dinvs[j] : 0.f;
    dadb[p] = f2bf(vv);
  }
  __syncthreads();

  // phase 5: gcn = relu(dad @ xw) * exp(v^2 * day) -> zin (one tile per wave)
  {
    const int mt = wave >> 2, ot = wave & 3;
    v8f acc = zero8();
    acc = wmma_bf16(load_a_frag(dadb, 32, mt*16, 0, lane),
                    load_b_frag(xwT, 32, ot*16, 0, lane), acc);
    const float dayf = (float)day_order[b];
#pragma unroll
    for (int i = 0; i < 8; ++i) {
      const int m = mt*16 + i + hi*8;
      if (m < NN) {
        const float fac = __expf(vvec[m]*vvec[m]*dayf);
        const float g = fmaxf(0.f, acc[i]) * fac;
        zin[((size_t)(m*BB + b)*DD + d)*65 + ot*16 + ln] = g;
      }
    }
  }
  if (tid < NN)
    zin[((size_t)(tid*BB + b)*DD + d)*65 + 64] = infection[(size_t)bd*NN + tid];
}

// ---------------------------------------------------------------------------
// Kernel 3: 2-layer LSTM encoder(28) + decoder(7) + fc head
// 4 waves/WG, each wave owns 16 batch rows; state in regs + per-wave LDS.
// ---------------------------------------------------------------------------
__global__ __launch_bounds__(128, 1)
void lstm_kernel(const float* __restrict__ zin, const float* __restrict__ infection,
                 const __bf16* __restrict__ wsb,
                 const float* __restrict__ b0, const float* __restrict__ b1,
                 const float* __restrict__ fc_w, const float* __restrict__ fc_b,
                 float* __restrict__ yout) {
  __shared__ __bf16 h0_buf[4][16*128];
  __shared__ __bf16 h1_buf[4][16*128];
  __shared__ __bf16 xm_buf[4][16*96];

  const int wave = threadIdx.x >> 5, lane = threadIdx.x & 31;
  const int hi = (lane >> 4) & 1, ln = lane & 15;
  const int rbase = blockIdx.x * 64 + wave * 16;
  __bf16* H0 = h0_buf[wave];
  __bf16* H1 = h1_buf[wave];
  __bf16* XM = xm_buf[wave];
  const __bf16* wih0b = wsb + OFF_WIH0;
  const __bf16* whh0b = wsb + OFF_WHH0;
  const __bf16* wih1b = wsb + OFF_WIH1;
  const __bf16* whh1b = wsb + OFF_WHH1;

  for (int i = lane; i < 16*128; i += 32) { H0[i] = f2bf(0.f); H1[i] = f2bf(0.f); }

  v8f c0[8], c1[8];
#pragma unroll
  for (int hb = 0; hb < 8; ++hb) { c0[hb] = zero8(); c1[hb] = zero8(); }

  // fc head: constant infection-feature dot per row (lanes 0..15 each own a row)
  float infdot = 0.f;
  const int rrow = rbase + ln;
  const int nidx = rrow / BB, bidx = rrow % BB;
  if (lane < 16) {
    float s = fc_b[0];
    for (int dd2 = 0; dd2 < DD; ++dd2)
      s += infection[((size_t)bidx*DD + dd2)*NN + nidx] * fc_w[HH + dd2];
    infdot = s;
  }
  __syncthreads();

  for (int step = 0; step < DD + 7; ++step) {
    const bool enc = step < DD;
    if (step == DD) {  // build decoder input = mean over encoder steps
      for (int idx = lane; idx < 16*96; idx += 32) {
        const int r = idx / 96, k = idx % 96;
        float s = 0.f;
        if (k < 65) {
          const float* p = zin + ((size_t)(rbase + r)*DD)*65 + k;
          for (int dd2 = 0; dd2 < DD; ++dd2) s += p[dd2*65];
          s *= (1.f/28.f);
        }
        XM[idx] = f2bf(s);
      }
      __syncthreads();
    }

    // layer-0 input fragments (K = 96, padded)
    v16bf xf0, xf1, xf2;
    if (enc) {
      xf0 = load_x_frag(zin, rbase, step, 0, lane);
      xf1 = load_x_frag(zin, rbase, step, 32, lane);
      xf2 = load_x_frag(zin, rbase, step, 64, lane);
    } else {
      xf0 = load_a_frag(XM, 96, 0, 0, lane);
      xf1 = load_a_frag(XM, 96, 0, 32, lane);
      xf2 = load_a_frag(XM, 96, 0, 64, lane);
    }
    v16bf h0f[4], h1f[4];
#pragma unroll
    for (int kt = 0; kt < 4; ++kt) {
      h0f[kt] = load_a_frag(H0, 128, 0, kt*32, lane);
      h1f[kt] = load_a_frag(H1, 128, 0, kt*32, lane);
    }

    // ---- layer 0: gates = x@wih0^T + h0@whh0^T + b0 ----
#pragma unroll 1
    for (int hb = 0; hb < 8; ++hb) {
      v8f g[4];
#pragma unroll
      for (int p = 0; p < 4; ++p) {
        const int n0 = p*128 + hb*16;
        v8f acc = zero8();
        acc = wmma_bf16(xf0, load_b_frag(wih0b, 96, n0, 0, lane), acc);
        acc = wmma_bf16(xf1, load_b_frag(wih0b, 96, n0, 32, lane), acc);
        acc = wmma_bf16(xf2, load_b_frag(wih0b, 96, n0, 64, lane), acc);
#pragma unroll
        for (int kt = 0; kt < 4; ++kt)
          acc = wmma_bf16(h0f[kt], load_b_frag(whh0b, 128, n0, kt*32, lane), acc);
        const float bn = b0[n0 + ln];
#pragma unroll
        for (int i = 0; i < 8; ++i) acc[i] += bn;
        g[p] = acc;
      }
#pragma unroll
      for (int i = 0; i < 8; ++i) {
        const float ig = sigm(g[0][i]), fg = sigm(g[1][i]);
        const float gg = tanhf(g[2][i]), og = sigm(g[3][i]);
        const float cv = fg*c0[hb][i] + ig*gg;
        c0[hb][i] = cv;
        H0[(i + hi*8)*128 + hb*16 + ln] = f2bf(og * tanhf(cv));
      }
    }
    __syncthreads();

    v16bf x1f[4];
#pragma unroll
    for (int kt = 0; kt < 4; ++kt) x1f[kt] = load_a_frag(H0, 128, 0, kt*32, lane);

    // ---- layer 1: gates = h0new@wih1^T + h1@whh1^T + b1 ----
#pragma unroll 1
    for (int hb = 0; hb < 8; ++hb) {
      v8f g[4];
#pragma unroll
      for (int p = 0; p < 4; ++p) {
        const int n0 = p*128 + hb*16;
        v8f acc = zero8();
#pragma unroll
        for (int kt = 0; kt < 4; ++kt) {
          acc = wmma_bf16(x1f[kt], load_b_frag(wih1b, 128, n0, kt*32, lane), acc);
          acc = wmma_bf16(h1f[kt], load_b_frag(whh1b, 128, n0, kt*32, lane), acc);
        }
        const float bn = b1[n0 + ln];
#pragma unroll
        for (int i = 0; i < 8; ++i) acc[i] += bn;
        g[p] = acc;
      }
#pragma unroll
      for (int i = 0; i < 8; ++i) {
        const float ig = sigm(g[0][i]), fg = sigm(g[1][i]);
        const float gg = tanhf(g[2][i]), og = sigm(g[3][i]);
        const float cv = fg*c1[hb][i] + ig*gg;
        c1[hb][i] = cv;
        H1[(i + hi*8)*128 + hb*16 + ln] = f2bf(og * tanhf(cv));
      }
    }
    __syncthreads();

    // decoder: fc head  y = relu(h1 . fc_w[0:128] + infdot)
    if (!enc && lane < 16) {
      float s = infdot;
      for (int k = 0; k < HH; ++k) s += (float)H1[ln*128 + k] * fc_w[k];
      yout[(size_t)bidx*(7*NN) + (size_t)(step - DD)*NN + nidx] = fmaxf(0.f, s);
    }
    __syncthreads();
  }
}

// ---------------------------------------------------------------------------
extern "C" void kernel_launch(void* const* d_in, const int* in_sizes, int n_in,
                              void* d_out, int out_size, void* d_ws, size_t ws_size,
                              hipStream_t stream) {
  (void)in_sizes; (void)n_in; (void)out_size; (void)ws_size;
  const float* x         = (const float*)d_in[0];
  const float* adj_real  = (const float*)d_in[1];
  const float* infection = (const float*)d_in[2];
  const int*   day_order = (const int*)  d_in[3];
  const float* gl_w1     = (const float*)d_in[4];
  const float* gl_b1     = (const float*)d_in[5];
  const float* gl_w2     = (const float*)d_in[6];
  const float* gl_b2     = (const float*)d_in[7];
  const float* glp       = (const float*)d_in[8];
  const float* gcn_w     = (const float*)d_in[9];
  const float* vvec      = (const float*)d_in[10];
  const float* wih0      = (const float*)d_in[11];
  const float* whh0      = (const float*)d_in[12];
  const float* b0        = (const float*)d_in[13];
  const float* wih1      = (const float*)d_in[14];
  const float* whh1      = (const float*)d_in[15];
  const float* b1        = (const float*)d_in[16];
  const float* fc_w      = (const float*)d_in[17];
  const float* fc_b      = (const float*)d_in[18];

  float*  zin = (float*)d_ws;
  __bf16* wsb = (__bf16*)((char*)d_ws + ZIN_ELEMS*sizeof(float));
  float*  adj_out = (float*)d_out;
  float*  yout = adj_out + ADJ_ELEMS;

  convert_weights<<<(WB_TOTAL + 255)/256, 256, 0, stream>>>(
      gl_w1, gl_w2, gcn_w, wih0, whh0, wih1, whh1, wsb);
  graph_kernel<<<BB*DD, 256, 0, stream>>>(
      x, adj_real, infection, day_order, gl_b1, gl_b2, glp, vvec, wsb, zin, adj_out);
  lstm_kernel<<<NB/64, 128, 0, stream>>>(
      zin, infection, wsb, b0, b1, fc_w, fc_b, yout);
}
